// CAM_Module_12670153523442
// MI455X (gfx1250) — matrix operand order repaired and checked by reading, not compiled, not run on hardware
//
#include <hip/hip_runtime.h>
#include <hip/hip_bf16.h>

// Problem constants: x is [B, C, H, W] = [32, 512, 48, 48] fp32
#define B_  32
#define C_  512
#define N_  2304   // 48*48

typedef __attribute__((ext_vector_type(16))) __bf16 v16bf;
typedef __attribute__((ext_vector_type(8)))  __bf16 v8bf;
typedef __attribute__((ext_vector_type(8)))  float  v8f;

// ---------------------------------------------------------------------------
// Kernel 1: convert x (f32) -> vbf [B,C,N] bf16 and vT [B,N,C] bf16 (transposed)
// 32x32 tiles through LDS so both writes are coalesced.
// ---------------------------------------------------------------------------
__global__ void __launch_bounds__(256)
convert_transpose_kernel(const float* __restrict__ x,
                         __hip_bfloat16* __restrict__ vbf,
                         __hip_bfloat16* __restrict__ vT) {
    __shared__ __hip_bfloat16 tile[32][33];
    const int b  = blockIdx.z;
    const int n0 = blockIdx.x * 32;
    const int c0 = blockIdx.y * 32;
    const int tx = threadIdx.x;   // 0..31
    const int ty = threadIdx.y;   // 0..7
#pragma unroll
    for (int i = 0; i < 4; ++i) {
        const int c = c0 + ty + 8 * i;
        const size_t idx = ((size_t)b * C_ + c) * N_ + n0 + tx;
        __hip_bfloat16 bv = __float2bfloat16(x[idx]);
        vbf[idx] = bv;
        tile[ty + 8 * i][tx] = bv;
    }
    __syncthreads();
#pragma unroll
    for (int i = 0; i < 4; ++i) {
        const int n = n0 + ty + 8 * i;
        vT[((size_t)b * N_ + n) * C_ + c0 + tx] = tile[tx][ty + 8 * i];
    }
}

// Build a 16x32 bf16 A-fragment from two contiguous 8-element chunks.
static __device__ __forceinline__ v16bf make_a_frag(const __hip_bfloat16* rowp,
                                                    int k, int khalf) {
    v8bf lo = *(const v8bf*)(rowp + k + khalf * 16);
    v8bf hi = *(const v8bf*)(rowp + k + khalf * 16 + 8);
    return __builtin_shufflevector(lo, hi, 0, 1, 2, 3, 4, 5, 6, 7,
                                           8, 9, 10, 11, 12, 13, 14, 15);
}

// ---------------------------------------------------------------------------
// Kernel 2: energy[b] = V[b] * V[b]^T   (C x C, f32 accum via bf16 WMMA)
// Block: 256 thr = 8 waves (2 along M x 4 along J); block tile 64(M) x 256(J).
// Wave tile 32 x 64 = 2x4 fragments (8 accumulators) -> 1.5 b128 loads / WMMA.
// ---------------------------------------------------------------------------
__global__ void __launch_bounds__(256)
energy_gemm_kernel(const __hip_bfloat16* __restrict__ vbf,
                   float* __restrict__ energy) {
    const int b    = blockIdx.z;
    const int lane = threadIdx.x & 31;
    const int wave = threadIdx.x >> 5;   // 0..7
    const int wm   = wave & 1;           // 2 waves along M
    const int wn   = wave >> 1;          // 4 waves along J
    const int m0   = blockIdx.y * 64 + wm * 32;
    const int j0   = blockIdx.x * 256 + wn * 64;
    const int lrow  = lane & 15;
    const int khalf = lane >> 4;

    const __hip_bfloat16* vb = vbf + (size_t)b * C_ * N_;
    const __hip_bfloat16* arow[2];
    arow[0] = vb + (size_t)(m0 + lrow) * N_;
    arow[1] = arow[0] + (size_t)16 * N_;
    const __hip_bfloat16* brow[4];
    brow[0] = vb + (size_t)(j0 + lrow) * N_;
    brow[1] = brow[0] + (size_t)16 * N_;
    brow[2] = brow[0] + (size_t)32 * N_;
    brow[3] = brow[0] + (size_t)48 * N_;

    v8f acc[2][4];
#pragma unroll
    for (int mi = 0; mi < 2; ++mi)
#pragma unroll
        for (int ji = 0; ji < 4; ++ji)
            acc[mi][ji] = (v8f){};

    for (int k = 0; k < N_; k += 32) {
        __builtin_prefetch(arow[0] + k + 256, 0, 3);  // WGP-scope prefetch
        __builtin_prefetch(brow[0] + k + 256, 0, 3);
        v16bf a[2];
#pragma unroll
        for (int mi = 0; mi < 2; ++mi)
            a[mi] = make_a_frag(arow[mi], k, khalf);
        v16bf bf[4];
#pragma unroll
        for (int ji = 0; ji < 4; ++ji)
            bf[ji] = *(const v16bf*)(brow[ji] + k + khalf * 16);
#pragma unroll
        for (int mi = 0; mi < 2; ++mi)
#pragma unroll
            for (int ji = 0; ji < 4; ++ji)
                acc[mi][ji] = __builtin_amdgcn_wmma_f32_16x16x32_bf16(
                    false, a[mi], false, bf[ji], (short)0, acc[mi][ji], false, false);
    }

    float* Eb = energy + (size_t)b * C_ * C_;
    const int rbase = 8 * khalf;
#pragma unroll
    for (int mi = 0; mi < 2; ++mi) {
#pragma unroll
        for (int r = 0; r < 8; ++r) {
            const int row = m0 + mi * 16 + r + rbase;
#pragma unroll
            for (int ji = 0; ji < 4; ++ji) {
                const int col = j0 + ji * 16 + lrow;
                Eb[(size_t)row * C_ + col] = acc[mi][ji][r];
            }
        }
    }
}

// ---------------------------------------------------------------------------
// Kernel 3: negated softmax. softmax(rowmax - e) == exp(rowmin - e)/sum.
// Store UNNORMALIZED p as bf16 + per-row f32 sum (normalization folded into
// GEMM2's f32 epilogue for better precision).
// One block of 512 threads per (b, row).
// ---------------------------------------------------------------------------
__global__ void __launch_bounds__(512)
softmax_kernel(const float* __restrict__ energy,
               __hip_bfloat16* __restrict__ P,
               float* __restrict__ rowsum) {
    __shared__ float red[16];
    const int row = blockIdx.x;                 // 0 .. B*C-1
    const float* e = energy + (size_t)row * C_;
    const int t   = threadIdx.x;
    const int wid = t >> 5;
    const int lid = t & 31;

    const float v = e[t];

    // row min (wave32 shuffles + LDS across 16 waves)
    float mn = v;
#pragma unroll
    for (int off = 16; off > 0; off >>= 1)
        mn = fminf(mn, __shfl_down(mn, off, 32));
    if (lid == 0) red[wid] = mn;
    __syncthreads();
    if (t < 16) {
        float m2 = red[t];
#pragma unroll
        for (int off = 8; off > 0; off >>= 1)
            m2 = fminf(m2, __shfl_down(m2, off, 32));
        if (t == 0) red[0] = m2;
    }
    __syncthreads();
    const float minv = red[0];
    __syncthreads();

    const float p = expf(minv - v);   // in (0, 1]

    float s = p;
#pragma unroll
    for (int off = 16; off > 0; off >>= 1)
        s += __shfl_down(s, off, 32);
    if (lid == 0) red[wid] = s;
    __syncthreads();
    if (t < 16) {
        float s2 = red[t];
#pragma unroll
        for (int off = 8; off > 0; off >>= 1)
            s2 += __shfl_down(s2, off, 32);
        if (t == 0) rowsum[row] = s2;
    }

    P[(size_t)row * C_ + t] = __float2bfloat16(p);
}

// ---------------------------------------------------------------------------
// Kernel 4: out = gamma * (P @ V) / rowsum + x
// A = P [C,C] bf16 row-major; B from vT [N,C] bf16 (K contiguous per lane).
// Same 64x256 block / 32x64 wave tiling. Fused f32 epilogue.
// ---------------------------------------------------------------------------
__global__ void __launch_bounds__(256)
out_gemm_kernel(const __hip_bfloat16* __restrict__ P,
                const __hip_bfloat16* __restrict__ vT,
                const float* __restrict__ rowsum,
                const float* __restrict__ x,
                const float* __restrict__ gamma,
                float* __restrict__ out) {
    const int b    = blockIdx.z;
    const int lane = threadIdx.x & 31;
    const int wave = threadIdx.x >> 5;
    const int wm   = wave & 1;
    const int wn   = wave >> 1;
    const int m0   = blockIdx.y * 64 + wm * 32;    // channel rows
    const int n0   = blockIdx.x * 256 + wn * 64;   // spatial cols
    const int lrow  = lane & 15;
    const int khalf = lane >> 4;

    const __hip_bfloat16* Pb = P + (size_t)b * C_ * C_;
    const __hip_bfloat16* arow[2];
    arow[0] = Pb + (size_t)(m0 + lrow) * C_;
    arow[1] = arow[0] + (size_t)16 * C_;
    const __hip_bfloat16* vTb = vT + (size_t)b * N_ * C_;
    const __hip_bfloat16* brow[4];
    brow[0] = vTb + (size_t)(n0 + lrow) * C_;
    brow[1] = brow[0] + (size_t)16 * C_;
    brow[2] = brow[0] + (size_t)32 * C_;
    brow[3] = brow[0] + (size_t)48 * C_;

    v8f acc[2][4];
#pragma unroll
    for (int mi = 0; mi < 2; ++mi)
#pragma unroll
        for (int ji = 0; ji < 4; ++ji)
            acc[mi][ji] = (v8f){};

    for (int k = 0; k < C_; k += 32) {
        __builtin_prefetch(brow[0] + k + 256, 0, 3);  // WGP-scope prefetch
        v16bf a[2];
#pragma unroll
        for (int mi = 0; mi < 2; ++mi)
            a[mi] = make_a_frag(arow[mi], k, khalf);
        v16bf bf[4];
#pragma unroll
        for (int ji = 0; ji < 4; ++ji)
            bf[ji] = *(const v16bf*)(brow[ji] + k + khalf * 16);
#pragma unroll
        for (int mi = 0; mi < 2; ++mi)
#pragma unroll
            for (int ji = 0; ji < 4; ++ji)
                acc[mi][ji] = __builtin_amdgcn_wmma_f32_16x16x32_bf16(
                    false, a[mi], false, bf[ji], (short)0, acc[mi][ji], false, false);
    }

    const float g = gamma[0];
    const float* xb = x + (size_t)b * C_ * N_;
    float* ob = out + (size_t)b * C_ * N_;
    const int rbase = 8 * khalf;
#pragma unroll
    for (int mi = 0; mi < 2; ++mi) {
#pragma unroll
        for (int r = 0; r < 8; ++r) {
            const int row = m0 + mi * 16 + r + rbase;
            const float sc = g / rowsum[b * C_ + row];
#pragma unroll
            for (int ji = 0; ji < 4; ++ji) {
                const int col = n0 + ji * 16 + lrow;
                const size_t idx = (size_t)row * N_ + col;
                ob[idx] = acc[mi][ji][r] * sc + xb[idx];
            }
        }
    }
}

// ---------------------------------------------------------------------------
extern "C" void kernel_launch(void* const* d_in, const int* in_sizes, int n_in,
                              void* d_out, int out_size, void* d_ws, size_t ws_size,
                              hipStream_t stream) {
    const float* x     = (const float*)d_in[0];
    const float* gamma = (const float*)d_in[1];
    float* out = (float*)d_out;

    // Workspace layout (all offsets 1KB-aligned):
    char* ws = (char*)d_ws;
    size_t off = 0;
    __hip_bfloat16* vbf = (__hip_bfloat16*)(ws + off); off += (size_t)B_ * C_ * N_ * 2; // 75.5 MB
    __hip_bfloat16* vT  = (__hip_bfloat16*)(ws + off); off += (size_t)B_ * C_ * N_ * 2; // 75.5 MB
    float* energy       = (float*)(ws + off);          off += (size_t)B_ * C_ * C_ * 4; // 33.6 MB
    __hip_bfloat16* P   = (__hip_bfloat16*)(ws + off); off += (size_t)B_ * C_ * C_ * 2; // 16.8 MB
    float* rowsum       = (float*)(ws + off);          off += (size_t)B_ * C_ * 4;      // 64 KB

    // 1) f32 -> bf16 (row-major + transposed copies)
    convert_transpose_kernel<<<dim3(N_ / 32, C_ / 32, B_), dim3(32, 8), 0, stream>>>(
        x, vbf, vT);

    // 2) energy = V V^T  (bf16 WMMA, f32 accum)
    energy_gemm_kernel<<<dim3(C_ / 256, C_ / 64, B_), 256, 0, stream>>>(vbf, energy);

    // 3) negated softmax -> unnormalized bf16 P + f32 row sums
    softmax_kernel<<<dim3(B_ * C_), 512, 0, stream>>>(energy, P, rowsum);

    // 4) out = gamma * (P V) / rowsum + x  (bf16 WMMA, fused f32 epilogue)
    out_gemm_kernel<<<dim3(N_ / 256, C_ / 64, B_), 256, 0, stream>>>(
        P, vT, rowsum, x, gamma, out);
}
